// TritterMLP_27573690040806
// MI455X (gfx1250) — compile-verified, matrix operand order checked by reading, not run on hardware
//
#include <hip/hip_runtime.h>
#include <hip/hip_bf16.h>

// Problem dims (compile-time constants from the reference)
#define HIDDEN 4096
#define INTER  11008
#define NTOK   4096   // BATCH * SEQ = 2 * 2048

// LDS staging: A tile = 128 rows x 64 K (bf16) per stage, double buffered.
// Row pitch 144 B = 128 B data + 16 B pad -> bank stride 36, conflict-free
// ds_load_b128 across lanes 0-15 (36*r mod 64 covers all 16 4-bank groups).
#define LDS_PITCH   144
#define STAGE_BYTES (128 * LDS_PITCH)   // 18432 B, x2 buffers = 36 KB

typedef __attribute__((ext_vector_type(16))) __bf16         v16bf;
typedef __attribute__((ext_vector_type(8)))  float          v8f;
typedef __attribute__((ext_vector_type(8)))  unsigned short u16x8;

// A/B fragment: 16 bf16 per lane = two 16-byte chunks.
// ISA 16-bit A-matrix 16x32 layout: lanes 0-15 hold K=0..7 (v0..v3) and
// K=16..23 (v4..v7); lanes 16-31 hold K=8..15 and K=24..31. Both chunks are
// contiguous along K -> b128 loads (global or ds).
union FragU {
  u16x8 h[2];
  v16bf v;
};

__device__ __forceinline__ unsigned short f32_to_bf16(float f) {
  unsigned int u = __float_as_uint(f);
  u += 0x7FFFu + ((u >> 16) & 1u);   // round-to-nearest-even
  return (unsigned short)(u >> 16);
}

__device__ __forceinline__ float silu_f(float g) {
  return g / (1.0f + __expf(-g));
}

// Async DMA: global (16B) -> LDS, bypasses VGPRs, tracked by ASYNCcnt.
__device__ __forceinline__ void async_g2lds_b128(unsigned lds_off,
                                                 const unsigned short* g) {
  asm volatile("global_load_async_to_lds_b128 %0, %1, off"
               :
               : "v"(lds_off), "v"(g)
               : "memory");
}

__device__ __forceinline__ void wait_async_le4() {
  asm volatile("s_wait_asynccnt 0x4" ::: "memory");
}
__device__ __forceinline__ void wait_async_le0() {
  asm volatile("s_wait_asynccnt 0x0" ::: "memory");
}

// ---------------------------------------------------------------------------
// f32 -> bf16 conversion (8 elements / thread, b128 load + b128 store)
// ---------------------------------------------------------------------------
__global__ void cvt_f32_to_bf16_kernel(const float* __restrict__ in,
                                       unsigned short* __restrict__ out,
                                       size_t n8) {
  size_t i = (size_t)blockIdx.x * blockDim.x + threadIdx.x;
  if (i >= n8) return;
  size_t base = i * 8;
  const float4* p = (const float4*)(in + base);
  float4 a = p[0];
  float4 b = p[1];
  u16x8 o;
  o[0] = f32_to_bf16(a.x); o[1] = f32_to_bf16(a.y);
  o[2] = f32_to_bf16(a.z); o[3] = f32_to_bf16(a.w);
  o[4] = f32_to_bf16(b.x); o[5] = f32_to_bf16(b.y);
  o[6] = f32_to_bf16(b.z); o[7] = f32_to_bf16(b.w);
  *(u16x8*)(out + base) = o;
}

// ---------------------------------------------------------------------------
// GEMM 1 (fused gate + up):
//   G = X * Wg^T + bg ; U = X * Wu^T + bu ; H = silu(G) * U  (bf16 out)
// A: NTOK x HIDDEN bf16 (staged in LDS), Bg/Bu: INTER x HIDDEN bf16 (direct).
// WG = 256 threads = 8 waves in 2(M) x 4(N); wave tile 64x32; WG tile 128x128.
// ---------------------------------------------------------------------------
__global__ __launch_bounds__(256) void swiglu_gemm1_kernel(
    const unsigned short* __restrict__ A,
    const unsigned short* __restrict__ Bg,
    const unsigned short* __restrict__ Bu,
    const float* __restrict__ bias_g,
    const float* __restrict__ bias_u,
    unsigned short* __restrict__ H) {
  const int K = HIDDEN;
  const int N = INTER;

  __shared__ __align__(16) unsigned char smem[2 * STAGE_BYTES];
  const unsigned smem_base = (unsigned)(unsigned long long)&smem[0];

  const int tid  = threadIdx.x;
  const int lane = tid & 31;
  const int wave = tid >> 5;
  const int hgrp = lane >> 4;   // 0: lanes 0-15, 1: lanes 16-31
  const int lid  = lane & 15;

  const int mBlk = blockIdx.y * 128;                 // WG M base (LDS tile rows)
  const int m0w  = (wave >> 2) * 64;                 // wave M offset in tile
  const int n0   = blockIdx.x * 128 + (wave & 3) * 32;

  // B pointers: weights row n, contiguous along K.
  const unsigned short* pBg[2];
  const unsigned short* pBu[2];
#pragma unroll
  for (int ni = 0; ni < 2; ++ni) {
    size_t off = (size_t)(n0 + ni * 16 + lid) * K + hgrp * 8;
    pBg[ni] = Bg + off;
    pBu[ni] = Bu + off;
  }

  v8f accg[4][2];
  v8f accu[4][2];
  const v8f vzero = {0.f, 0.f, 0.f, 0.f, 0.f, 0.f, 0.f, 0.f};
#pragma unroll
  for (int mi = 0; mi < 4; ++mi)
#pragma unroll
    for (int ni = 0; ni < 2; ++ni) {
      accg[mi][ni] = vzero;
      accu[mi][ni] = vzero;
    }

  // Async-stage the 128x64 A tile for stage s into buffer (s&1).
  // 1024 16B chunks / 256 threads = 4 async b128 DMAs per thread.
  auto stage = [&](int s) {
    const unsigned short* gbase = A + (size_t)mBlk * K + (size_t)s * 64;
    const unsigned lbase = smem_base + (unsigned)((s & 1) * STAGE_BYTES);
#pragma unroll
    for (int i = 0; i < 4; ++i) {
      int c   = tid + i * 256;
      int row = c >> 3;
      int col = c & 7;
      async_g2lds_b128(lbase + (unsigned)(row * LDS_PITCH + col * 16),
                       gbase + (size_t)row * K + col * 8);
    }
  };

  const int S = K / 64;   // 64 stages
  stage(0);

  for (int s = 0; s < S; ++s) {
    if (s + 1 < S) {
      stage(s + 1);
      wait_async_le4();   // in-order async: my stage(s) chunks are in LDS
    } else {
      wait_async_le0();
    }
    __syncthreads();      // everyone's stage(s) chunks are in LDS

    const unsigned char* lbuf = &smem[(s & 1) * STAGE_BYTES];
#pragma unroll
    for (int hs = 0; hs < 2; ++hs) {
      const int ks = hs * 32;           // K offset inside stage
      const int kg = s * 64 + ks;       // global K offset

      FragU a[4], bg[2], bu[2];
#pragma unroll
      for (int mi = 0; mi < 4; ++mi) {
        const unsigned char* p =
            lbuf + (m0w + mi * 16 + lid) * LDS_PITCH + ks * 2 + hgrp * 16;
        a[mi].h[0] = *(const u16x8*)(p);
        a[mi].h[1] = *(const u16x8*)(p + 32);
      }
#pragma unroll
      for (int ni = 0; ni < 2; ++ni) {
        bg[ni].h[0] = *(const u16x8*)(pBg[ni] + kg);
        bg[ni].h[1] = *(const u16x8*)(pBg[ni] + kg + 16);
        bu[ni].h[0] = *(const u16x8*)(pBu[ni] + kg);
        bu[ni].h[1] = *(const u16x8*)(pBu[ni] + kg + 16);
      }
#pragma unroll
      for (int mi = 0; mi < 4; ++mi)
#pragma unroll
        for (int ni = 0; ni < 2; ++ni) {
          accg[mi][ni] = __builtin_amdgcn_wmma_f32_16x16x32_bf16(
              false, a[mi].v, false, bg[ni].v, (short)0, accg[mi][ni], false, false);
          accu[mi][ni] = __builtin_amdgcn_wmma_f32_16x16x32_bf16(
              false, a[mi].v, false, bu[ni].v, (short)0, accu[mi][ni], false, false);
        }
    }
    __syncthreads();      // all reads of buffer (s&1) done before s+2 overwrites
  }

  // Epilogue: C/D layout -> lanes 0-15: M=j, N=lane; lanes 16-31: M=j+8, N=lane-16
#pragma unroll
  for (int ni = 0; ni < 2; ++ni) {
    const int n = n0 + ni * 16 + lid;
    const float bgv = bias_g[n];
    const float buv = bias_u[n];
#pragma unroll
    for (int mi = 0; mi < 4; ++mi) {
#pragma unroll
      for (int j = 0; j < 8; ++j) {
        float g = accg[mi][ni][j] + bgv;
        float u = accu[mi][ni][j] + buv;
        int m = mBlk + m0w + mi * 16 + hgrp * 8 + j;
        H[(size_t)m * N + n] = f32_to_bf16(silu_f(g) * u);
      }
    }
  }
}

// ---------------------------------------------------------------------------
// GEMM 2 (down proj): OUT = H * Wd^T + bd   (f32 out)
// A: NTOK x INTER bf16 (staged in LDS), B: HIDDEN x INTER bf16 (direct).
// WG = 8 waves in 2(M) x 4(N); wave tile 64x64 (4x4 WMMA); WG tile 128x256.
// ---------------------------------------------------------------------------
__global__ __launch_bounds__(256) void down_gemm_kernel(
    const unsigned short* __restrict__ A,
    const unsigned short* __restrict__ B,
    const float* __restrict__ bias,
    float* __restrict__ OUT) {
  const int K = INTER;
  const int N = HIDDEN;

  __shared__ __align__(16) unsigned char smem[2 * STAGE_BYTES];
  const unsigned smem_base = (unsigned)(unsigned long long)&smem[0];

  const int tid  = threadIdx.x;
  const int lane = tid & 31;
  const int wave = tid >> 5;
  const int hgrp = lane >> 4;
  const int lid  = lane & 15;

  const int mBlk = blockIdx.y * 128;
  const int m0w  = (wave >> 2) * 64;
  const int n0   = blockIdx.x * 256 + (wave & 3) * 64;

  const unsigned short* pB[4];
#pragma unroll
  for (int ni = 0; ni < 4; ++ni)
    pB[ni] = B + (size_t)(n0 + ni * 16 + lid) * K + hgrp * 8;

  v8f acc[4][4];
  const v8f vzero = {0.f, 0.f, 0.f, 0.f, 0.f, 0.f, 0.f, 0.f};
#pragma unroll
  for (int mi = 0; mi < 4; ++mi)
#pragma unroll
    for (int ni = 0; ni < 4; ++ni) acc[mi][ni] = vzero;

  auto stage = [&](int s) {
    const unsigned short* gbase = A + (size_t)mBlk * K + (size_t)s * 64;
    const unsigned lbase = smem_base + (unsigned)((s & 1) * STAGE_BYTES);
#pragma unroll
    for (int i = 0; i < 4; ++i) {
      int c   = tid + i * 256;
      int row = c >> 3;
      int col = c & 7;
      async_g2lds_b128(lbase + (unsigned)(row * LDS_PITCH + col * 16),
                       gbase + (size_t)row * K + col * 8);
    }
  };

  const int S = K / 64;   // 172 stages
  stage(0);

  for (int s = 0; s < S; ++s) {
    if (s + 1 < S) {
      stage(s + 1);
      wait_async_le4();
    } else {
      wait_async_le0();
    }
    __syncthreads();

    const unsigned char* lbuf = &smem[(s & 1) * STAGE_BYTES];
#pragma unroll
    for (int hs = 0; hs < 2; ++hs) {
      const int ks = hs * 32;
      const int kg = s * 64 + ks;

      FragU a[4], b[4];
#pragma unroll
      for (int mi = 0; mi < 4; ++mi) {
        const unsigned char* p =
            lbuf + (m0w + mi * 16 + lid) * LDS_PITCH + ks * 2 + hgrp * 16;
        a[mi].h[0] = *(const u16x8*)(p);
        a[mi].h[1] = *(const u16x8*)(p + 32);
      }
#pragma unroll
      for (int ni = 0; ni < 4; ++ni) {
        b[ni].h[0] = *(const u16x8*)(pB[ni] + kg);
        b[ni].h[1] = *(const u16x8*)(pB[ni] + kg + 16);
      }
#pragma unroll
      for (int mi = 0; mi < 4; ++mi)
#pragma unroll
        for (int ni = 0; ni < 4; ++ni)
          acc[mi][ni] = __builtin_amdgcn_wmma_f32_16x16x32_bf16(
              false, a[mi].v, false, b[ni].v, (short)0, acc[mi][ni], false, false);
    }
    __syncthreads();
  }

#pragma unroll
  for (int ni = 0; ni < 4; ++ni) {
    const int n = n0 + ni * 16 + lid;
    const float bv = bias[n];
#pragma unroll
    for (int mi = 0; mi < 4; ++mi) {
#pragma unroll
      for (int j = 0; j < 8; ++j) {
        int m = mBlk + m0w + mi * 16 + hgrp * 8 + j;
        OUT[(size_t)m * N + n] = acc[mi][ni][j] + bv;
      }
    }
  }
}

// ---------------------------------------------------------------------------
// Launch: cvt(X,Wg,Wu) -> gemm1 -> cvt(Wd into Wg slot) -> gemm2
// Workspace (bf16 elems): Xb 16.8M (32MB) | W1 45.1M (86MB) | W2 45.1M (86MB)
//                         | Hb 45.1M (86MB)   => ~304 MB total
// ---------------------------------------------------------------------------
extern "C" void kernel_launch(void* const* d_in, const int* in_sizes, int n_in,
                              void* d_out, int out_size, void* d_ws, size_t ws_size,
                              hipStream_t stream) {
  const float* x  = (const float*)d_in[0];
  const float* wg = (const float*)d_in[1];
  const float* bg = (const float*)d_in[2];
  const float* wu = (const float*)d_in[3];
  const float* bu = (const float*)d_in[4];
  const float* wd = (const float*)d_in[5];
  const float* bd = (const float*)d_in[6];
  float* out = (float*)d_out;

  unsigned short* Xb = (unsigned short*)d_ws;
  unsigned short* W1 = Xb + (size_t)NTOK * HIDDEN;
  unsigned short* W2 = W1 + (size_t)INTER * HIDDEN;
  unsigned short* Hb = W2 + (size_t)INTER * HIDDEN;

  const size_t nX8 = (size_t)NTOK * HIDDEN / 8;
  const size_t nW8 = (size_t)INTER * HIDDEN / 8;

  cvt_f32_to_bf16_kernel<<<(unsigned)((nX8 + 255) / 256), 256, 0, stream>>>(x, Xb, nX8);
  cvt_f32_to_bf16_kernel<<<(unsigned)((nW8 + 255) / 256), 256, 0, stream>>>(wg, W1, nW8);
  cvt_f32_to_bf16_kernel<<<(unsigned)((nW8 + 255) / 256), 256, 0, stream>>>(wu, W2, nW8);

  swiglu_gemm1_kernel<<<dim3(INTER / 128, NTOK / 128), 256, 0, stream>>>(
      Xb, W1, W2, bg, bu, Hb);

  // W_gate no longer needed -> reuse its slot for W_down (stream-ordered).
  cvt_f32_to_bf16_kernel<<<(unsigned)((nW8 + 255) / 256), 256, 0, stream>>>(wd, W1, nW8);

  down_gemm_kernel<<<dim3(HIDDEN / 256, NTOK / 128), 256, 0, stream>>>(
      Hb, W1, bd, out);
}
